// ECForGraphTCN_21964462752072
// MI455X (gfx1250) — matrix operand order repaired and checked by reading, not compiled, not run on hardware
//
#include <hip/hip_runtime.h>
#include <hip/hip_bf16.h>
#include <math.h>

typedef __attribute__((ext_vector_type(16))) _Float16 v16h;
typedef __attribute__((ext_vector_type(8)))  float    v8f;

#define LD 64  // leading dim (in halves) of per-wave LDS activation tiles

// ---------------------------------------------------------------------------
// WMMA fragment builders (wave32; contiguous k order, same for A and B)
// A (16x32 f16, MxK): lanes 0-15 -> M=0..15 ; halves 0-15: K=0..15(+16 for hi
// lanes)
// ---------------------------------------------------------------------------
__device__ __forceinline__ v16h ld_afrag(const _Float16* act, int lane, int k0) {
  const int m = lane & 15;
  const int g = lane >> 4;
  v16h a;
#pragma unroll
  for (int h = 0; h < 16; ++h) {
    const int k = k0 + h + 16 * g;
    a[h] = act[m * LD + k];
  }
  return a;
}

// B (32x16 f16, KxN): lanes 0-15 hold K=0..15, lanes 16-31 hold K=16..31,
// N = lane & 15.  Weights stored row-major [Nout_pad][ldw] (W[n][k]), so
// B[k][n] = W[n][k] gives out = act @ W^T.
__device__ __forceinline__ v16h ld_bfrag(const _Float16* w, int ldw, int lane,
                                         int n0, int k0) {
  const int n = n0 + (lane & 15);
  const int g = lane >> 4;
  v16h b;
#pragma unroll
  for (int h = 0; h < 16; ++h) {
    b[h] = w[n * ldw + k0 + h + 16 * g];
  }
  return b;
}

// One linear layer on a 16-row tile: out[16][NT*16] = (relu?)(in @ W^T + b)
// C/D layout: VGPR r -> M = r + 8*(lane>=16), N = lane&15.
template <int NT, int KS, bool RELU>
__device__ __forceinline__ void mlp_layer(const _Float16* actIn, _Float16* actOut,
                                          const _Float16* sw, int ldw,
                                          const float* sb, int lane) {
  // make prior same-wave LDS stores visible before fragment gathers
  asm volatile("s_wait_dscnt 0x0" ::: "memory");
#pragma unroll
  for (int nt = 0; nt < NT; ++nt) {
    const float bv = sb[nt * 16 + (lane & 15)];
    v8f acc = {bv, bv, bv, bv, bv, bv, bv, bv};
#pragma unroll
    for (int ks = 0; ks < KS; ++ks) {
      v16h a = ld_afrag(actIn, lane, ks * 32);
      v16h b = ld_bfrag(sw, ldw, lane, nt * 16, ks * 32);
      acc = __builtin_amdgcn_wmma_f32_16x16x32_f16(false, a, false, b,
                                                   (short)0, acc, false, false);
    }
    const int n = nt * 16 + (lane & 15);
    const int g = lane >> 4;
#pragma unroll
    for (int r = 0; r < 8; ++r) {
      float v = acc[r];
      if (RELU) v = v > 0.f ? v : 0.f;
      actOut[(r + 8 * g) * LD + n] = (_Float16)v;
    }
  }
}

// ---------------------------------------------------------------------------
// Cooperative fp32->f16 weight/bias staging (zero padded)
// ---------------------------------------------------------------------------
__device__ __forceinline__ void stage_weights(const float* W, int dout, int din,
                                              _Float16* sw, int rows, int ldw,
                                              int tid, int nthr) {
  for (int i = tid; i < rows * ldw; i += nthr) {
    const int n = i / ldw;
    const int k = i - n * ldw;
    sw[i] = (_Float16)((n < dout && k < din) ? W[n * din + k] : 0.f);
  }
}
__device__ __forceinline__ void stage_bias(const float* B, int dout, float* sb,
                                           int rows, int tid, int nthr) {
  for (int i = tid; i < rows; i += nthr)
    sb[i] = (B != nullptr && i < dout) ? B[i] : 0.f;
}

__device__ __forceinline__ void zero_tiles(_Float16* a, _Float16* b, int lane) {
  for (int i = lane; i < 16 * LD; i += 32) {
    a[i] = (_Float16)0.f;
    b[i] = (_Float16)0.f;
  }
}

// ===========================================================================
// Node encoder: h = relu(relu(x @ W0^T) @ W1^T)    [7 -> 40 -> 5], no bias
// ===========================================================================
__global__ void ec_node_enc_kernel(const float* __restrict__ x,
                                   const float* __restrict__ W0,
                                   const float* __restrict__ W1,
                                   float* __restrict__ h, int nNodes) {
  __shared__ _Float16 sW1[48 * 32];
  __shared__ _Float16 sW2[16 * 64];
  __shared__ float sB1[48], sB2[16];
  __shared__ _Float16 sAct[8][2][16 * LD];

  const int tid = threadIdx.x, nthr = blockDim.x;
  stage_weights(W0, 40, 7, sW1, 48, 32, tid, nthr);
  stage_weights(W1, 5, 40, sW2, 16, 64, tid, nthr);
  stage_bias(nullptr, 0, sB1, 48, tid, nthr);
  stage_bias(nullptr, 0, sB2, 16, tid, nthr);
  __syncthreads();

  const int wave = tid >> 5, lane = tid & 31;
  const int n0 = (blockIdx.x * 8 + wave) * 16;
  if (n0 >= nNodes) return;

  _Float16* bufA = &sAct[wave][0][0];
  _Float16* bufB = &sAct[wave][1][0];
  zero_tiles(bufA, bufB, lane);

#pragma unroll
  for (int it = 0; it < 16; ++it) {
    const int node = min(n0 + it, nNodes - 1);
    if (lane < 7) bufA[it * LD + lane] = (_Float16)x[node * 7 + lane];
  }
  mlp_layer<3, 1, true>(bufA, bufB, sW1, 32, sB1, lane);  // 7 -> 40, relu
  mlp_layer<1, 2, true>(bufB, bufA, sW2, 64, sB2, lane);  // 40 -> 5, relu

  asm volatile("s_wait_dscnt 0x0" ::: "memory");
  for (int i = lane; i < 16 * 5; i += 32) {
    const int it = i / 5, c = i - it * 5;
    const int node = n0 + it;
    if (node < nNodes) h[node * 5 + c] = (float)bufA[it * LD + c];
  }
}

// ===========================================================================
// Edge encoder: e = relu(relu(ea @ W0^T) @ W1^T)   [4 -> 40 -> 4], no bias
// writes e_cat[:, 0:4]
// ===========================================================================
__global__ void ec_edge_enc_kernel(const float* __restrict__ ea,
                                   const float* __restrict__ W0,
                                   const float* __restrict__ W1,
                                   float* __restrict__ ecat, int nEdges) {
  __shared__ _Float16 sW1[48 * 32];
  __shared__ _Float16 sW2[16 * 64];
  __shared__ float sB1[48], sB2[16];
  __shared__ _Float16 sAct[8][2][16 * LD];

  const int tid = threadIdx.x, nthr = blockDim.x;
  stage_weights(W0, 40, 4, sW1, 48, 32, tid, nthr);
  stage_weights(W1, 4, 40, sW2, 16, 64, tid, nthr);
  stage_bias(nullptr, 0, sB1, 48, tid, nthr);
  stage_bias(nullptr, 0, sB2, 16, tid, nthr);
  __syncthreads();

  const int wave = tid >> 5, lane = tid & 31;
  const int e0 = (blockIdx.x * 8 + wave) * 16;
  if (e0 >= nEdges) return;

  _Float16* bufA = &sAct[wave][0][0];
  _Float16* bufB = &sAct[wave][1][0];
  zero_tiles(bufA, bufB, lane);

#pragma unroll
  for (int it = 0; it < 16; ++it) {
    const int e = min(e0 + it, nEdges - 1);
    if (lane < 4) bufA[it * LD + lane] = (_Float16)ea[e * 4 + lane];
  }
  mlp_layer<3, 1, true>(bufA, bufB, sW1, 32, sB1, lane);  // 4 -> 40, relu
  mlp_layer<1, 2, true>(bufB, bufA, sW2, 64, sB2, lane);  // 40 -> 4, relu

  asm volatile("s_wait_dscnt 0x0" ::: "memory");
  for (int i = lane; i < 16 * 4; i += 32) {
    const int it = i >> 2, c = i & 3;
    const int e = e0 + it;
    if (e < nEdges) ecat[(size_t)e * 16 + c] = (float)bufA[it * LD + c];
  }
}

__global__ void ec_zero_kernel(float* __restrict__ p, int n) {
  const int i = blockIdx.x * blockDim.x + threadIdx.x;
  if (i < n) p[i] = 0.f;
}

// ===========================================================================
// Relational model: e' = MLP([h[dst], h[src], e])  [14 -> 40 -> 40 -> 4]
// writes e_cat[:, 4*(layer+1)..] and scatter-adds e' into agg[dst]
// ===========================================================================
__global__ void ec_rel_kernel(const float* __restrict__ h,
                              const int* __restrict__ eidx,
                              float* __restrict__ ecat, float* __restrict__ agg,
                              const float* W0, const float* b0,
                              const float* W1, const float* b1,
                              const float* W2, const float* b2,
                              int nEdges, int layer) {
  __shared__ _Float16 sW1[48 * 32];
  __shared__ _Float16 sW2[48 * 64];
  __shared__ _Float16 sW3[16 * 64];
  __shared__ float sB1[48], sB2[48], sB3[16];
  __shared__ _Float16 sAct[8][2][16 * LD];

  const int tid = threadIdx.x, nthr = blockDim.x;
  stage_weights(W0, 40, 14, sW1, 48, 32, tid, nthr);
  stage_weights(W1, 40, 40, sW2, 48, 64, tid, nthr);
  stage_weights(W2, 4, 40, sW3, 16, 64, tid, nthr);
  stage_bias(b0, 40, sB1, 48, tid, nthr);
  stage_bias(b1, 40, sB2, 48, tid, nthr);
  stage_bias(b2, 4, sB3, 16, tid, nthr);
  __syncthreads();

  const int wave = tid >> 5, lane = tid & 31;
  const int e0 = (blockIdx.x * 8 + wave) * 16;
  if (e0 >= nEdges) return;

  _Float16* bufA = &sAct[wave][0][0];
  _Float16* bufB = &sAct[wave][1][0];
  zero_tiles(bufA, bufB, lane);

#pragma unroll
  for (int it = 0; it < 16; ++it) {
    const int e = min(e0 + it, nEdges - 1);
    const int src = eidx[e];
    const int dst = eidx[nEdges + e];
    const int k = lane;
    float v = 0.f;
    if (k < 5)        v = h[dst * 5 + k];          // x_i
    else if (k < 10)  v = h[src * 5 + (k - 5)];    // x_j
    else if (k < 14)  v = ecat[(size_t)e * 16 + 4 * layer + (k - 10)];
    if (k < 32) bufA[it * LD + k] = (_Float16)v;
  }
  mlp_layer<3, 1, true >(bufA, bufB, sW1, 32, sB1, lane);  // 14 -> 40
  mlp_layer<3, 2, true >(bufB, bufA, sW2, 64, sB2, lane);  // 40 -> 40
  mlp_layer<1, 2, false>(bufA, bufB, sW3, 64, sB3, lane);  // 40 -> 4

  asm volatile("s_wait_dscnt 0x0" ::: "memory");
  for (int i = lane; i < 16 * 4; i += 32) {
    const int it = i >> 2, c = i & 3;
    const int e = e0 + it;
    if (e < nEdges) {
      const float v = (float)bufB[it * LD + c];
      ecat[(size_t)e * 16 + 4 * (layer + 1) + c] = v;
      const int dst = eidx[nEdges + e];
      atomicAdd(&agg[dst * 4 + c], v);  // segment_sum -> global_atomic_add_f32 (L2)
    }
  }
}

// ===========================================================================
// Object model + residual: h = 0.5*h + 0.5*MLP([h, agg])  [9 -> 40 -> 40 -> 5]
// ===========================================================================
__global__ void ec_obj_kernel(float* __restrict__ h, const float* __restrict__ agg,
                              const float* W0, const float* b0,
                              const float* W1, const float* b1,
                              const float* W2, const float* b2, int nNodes) {
  __shared__ _Float16 sW1[48 * 32];
  __shared__ _Float16 sW2[48 * 64];
  __shared__ _Float16 sW3[16 * 64];
  __shared__ float sB1[48], sB2[48], sB3[16];
  __shared__ _Float16 sAct[8][2][16 * LD];

  const int tid = threadIdx.x, nthr = blockDim.x;
  stage_weights(W0, 40, 9, sW1, 48, 32, tid, nthr);
  stage_weights(W1, 40, 40, sW2, 48, 64, tid, nthr);
  stage_weights(W2, 5, 40, sW3, 16, 64, tid, nthr);
  stage_bias(b0, 40, sB1, 48, tid, nthr);
  stage_bias(b1, 40, sB2, 48, tid, nthr);
  stage_bias(b2, 5, sB3, 16, tid, nthr);
  __syncthreads();

  const int wave = tid >> 5, lane = tid & 31;
  const int n0 = (blockIdx.x * 8 + wave) * 16;
  if (n0 >= nNodes) return;

  _Float16* bufA = &sAct[wave][0][0];
  _Float16* bufB = &sAct[wave][1][0];
  zero_tiles(bufA, bufB, lane);

#pragma unroll
  for (int it = 0; it < 16; ++it) {
    const int node = min(n0 + it, nNodes - 1);
    const int k = lane;
    float v = 0.f;
    if (k < 5)      v = h[node * 5 + k];
    else if (k < 9) v = agg[node * 4 + (k - 5)];
    if (k < 32) bufA[it * LD + k] = (_Float16)v;
  }
  mlp_layer<3, 1, true >(bufA, bufB, sW1, 32, sB1, lane);  // 9 -> 40
  mlp_layer<3, 2, true >(bufB, bufA, sW2, 64, sB2, lane);  // 40 -> 40
  mlp_layer<1, 2, false>(bufA, bufB, sW3, 64, sB3, lane);  // 40 -> 5

  asm volatile("s_wait_dscnt 0x0" ::: "memory");
  for (int i = lane; i < 16 * 5; i += 32) {
    const int it = i / 5, c = i - it * 5;
    const int node = n0 + it;
    if (node < nNodes) {
      const float dh = (float)bufB[it * LD + c];
      h[node * 5 + c] = 0.5f * h[node * 5 + c] + 0.5f * dh;  // alpha = 0.5
    }
  }
}

// ===========================================================================
// Final classifier: sigmoid(MLP(e_cat))  [16 -> 40 -> 40 -> 1]
// ===========================================================================
__global__ void ec_final_kernel(const float* __restrict__ ecat,
                                float* __restrict__ out,
                                const float* W0, const float* b0,
                                const float* W1, const float* b1,
                                const float* W2, const float* b2, int nEdges) {
  __shared__ _Float16 sW1[48 * 32];
  __shared__ _Float16 sW2[48 * 64];
  __shared__ _Float16 sW3[16 * 64];
  __shared__ float sB1[48], sB2[48], sB3[16];
  __shared__ _Float16 sAct[8][2][16 * LD];

  const int tid = threadIdx.x, nthr = blockDim.x;
  stage_weights(W0, 40, 16, sW1, 48, 32, tid, nthr);
  stage_weights(W1, 40, 40, sW2, 48, 64, tid, nthr);
  stage_weights(W2, 1, 40, sW3, 16, 64, tid, nthr);
  stage_bias(b0, 40, sB1, 48, tid, nthr);
  stage_bias(b1, 40, sB2, 48, tid, nthr);
  stage_bias(b2, 1, sB3, 16, tid, nthr);
  __syncthreads();

  const int wave = tid >> 5, lane = tid & 31;
  const int e0 = (blockIdx.x * 8 + wave) * 16;
  if (e0 >= nEdges) return;

  _Float16* bufA = &sAct[wave][0][0];
  _Float16* bufB = &sAct[wave][1][0];
  zero_tiles(bufA, bufB, lane);

#pragma unroll
  for (int it = 0; it < 16; ++it) {
    const int e = min(e0 + it, nEdges - 1);
    if (lane < 16) bufA[it * LD + lane] = (_Float16)ecat[(size_t)e * 16 + lane];
  }
  mlp_layer<3, 1, true >(bufA, bufB, sW1, 32, sB1, lane);  // 16 -> 40
  mlp_layer<3, 2, true >(bufB, bufA, sW2, 64, sB2, lane);  // 40 -> 40
  mlp_layer<1, 2, false>(bufA, bufB, sW3, 64, sB3, lane);  // 40 -> 1

  asm volatile("s_wait_dscnt 0x0" ::: "memory");
  if (lane < 16) {
    const int e = e0 + lane;
    if (e < nEdges) {
      const float v = (float)bufB[lane * LD + 0];
      out[e] = 1.f / (1.f + __expf(-v));
    }
  }
}

// ===========================================================================
// Launch: pytree flatten order (sorted dict keys):
//  0:x  1:edge_index  2:edge_attr
//  3..8  : "W"        (W0.W, W0.b, W1.W, W1.b, W2.W, W2.b)
//  9..10 : "edge_enc" (W0, W1)
// 11..46 : "in_layers" i=0..2 : obj{W0,b0,W1,b1,W2,b2}, rel{W0,b0,W1,b1,W2,b2}
// 47..48 : "node_enc" (W0, W1)
// ===========================================================================
extern "C" void kernel_launch(void* const* d_in, const int* in_sizes, int n_in,
                              void* d_out, int out_size, void* d_ws, size_t ws_size,
                              hipStream_t stream) {
  const float* x     = (const float*)d_in[0];
  const int*   eidx  = (const int*)d_in[1];
  const float* eattr = (const float*)d_in[2];

  const float* fcW0 = (const float*)d_in[3];
  const float* fcB0 = (const float*)d_in[4];
  const float* fcW1 = (const float*)d_in[5];
  const float* fcB1 = (const float*)d_in[6];
  const float* fcW2 = (const float*)d_in[7];
  const float* fcB2 = (const float*)d_in[8];
  const float* eeW0 = (const float*)d_in[9];
  const float* eeW1 = (const float*)d_in[10];
  const float* neW0 = (const float*)d_in[47];
  const float* neW1 = (const float*)d_in[48];

  const int N = in_sizes[0] / 7;
  const int E = in_sizes[2] / 4;

  float* h    = (float*)d_ws;                 // N x 5
  float* agg  = h + (size_t)N * 5;            // N x 4
  float* ecat = agg + (size_t)N * 4;          // E x 16

  const int nodeBlocks = ((N + 15) / 16 + 7) / 8;
  const int edgeBlocks = ((E + 15) / 16 + 7) / 8;

  ec_node_enc_kernel<<<nodeBlocks, 256, 0, stream>>>(x, neW0, neW1, h, N);
  ec_edge_enc_kernel<<<edgeBlocks, 256, 0, stream>>>(eattr, eeW0, eeW1, ecat, E);

  for (int L = 0; L < 3; ++L) {
    const float* const* p = (const float* const*)(d_in + 11 + 12 * L);
    // p[0..5] = obj layer, p[6..11] = rel layer
    ec_zero_kernel<<<(N * 4 + 255) / 256, 256, 0, stream>>>(agg, N * 4);
    ec_rel_kernel<<<edgeBlocks, 256, 0, stream>>>(h, eidx, ecat, agg,
        p[6], p[7], p[8], p[9], p[10], p[11], E, L);
    ec_obj_kernel<<<nodeBlocks, 256, 0, stream>>>(h, agg,
        p[0], p[1], p[2], p[3], p[4], p[5], N);
  }

  ec_final_kernel<<<edgeBlocks, 256, 0, stream>>>(ecat, (float*)d_out,
      fcW0, fcB0, fcW1, fcB1, fcW2, fcB2, E);
}
